// LSTM_Model_36034775613965
// MI455X (gfx1250) — matrix operand order repaired and checked by reading, not compiled, not run on hardware
//
#include <hip/hip_runtime.h>

// ---------------- problem constants ----------------
#define BB   256          // batch
#define TT   512          // timesteps
#define DD   512          // input dim
#define HD   1024         // hidden dim
#define G4   (4 * HD)     // 4096 gate width

typedef __bf16 bf16;
typedef __attribute__((ext_vector_type(16))) __bf16 v16bf;
typedef __attribute__((ext_vector_type(8)))  __bf16 v8bf;
typedef __attribute__((ext_vector_type(8)))  float  v8f;

typedef __attribute__((address_space(3))) bf16 lds_bf16;

// ---------------------------------------------------------------------------
// CDNA5 async direct-to-LDS copy (VGLOBAL opcode 98, tracked by ASYNCcnt).
// Each active lane moves 16 bytes: global memory -> LDS at the lane's LDS offset.
// ---------------------------------------------------------------------------
__device__ __forceinline__ void async_load_b128_to_lds(const bf16* gaddr, bf16* lptr) {
  unsigned loff = (unsigned)(uintptr_t)(lds_bf16*)lptr;   // 32-bit LDS offset
  asm volatile("global_load_async_to_lds_b128 %0, %1, off"
               :: "v"(loff), "v"(gaddr)
               : "memory");
}
__device__ __forceinline__ void wait_asynccnt0() {
  asm volatile("s_wait_asynccnt 0x0" ::: "memory");
}

// ---------------------------------------------------------------------------
// WMMA fragment loaders (per CDNA5 ISA 7.12.2 layouts, wave32)
//
// A (16x32 bf16, row-major source, stride lda):
//   lanes 0-15  : row M=lane,    K = {k0+0..7, k0+16..23}
//   lanes 16-31 : row M=lane-16, K = {k0+8..15, k0+24..31}
// B (32x16 bf16) from BT = B^T stored row-major [N][K], stride ldb:
//   lanes 0-15  : col N=lane,    K = {k0+0..7, k0+16..23}
//   lanes 16-31 : col N=lane-16, K = {k0+8..15, k0+24..31}
// ---------------------------------------------------------------------------
__device__ __forceinline__ v16bf load_a_frag(const bf16* __restrict__ A, int lda,
                                             int m0, int k0, int lane) {
  const int row  = lane & 15;
  const int koff = (lane & 16) ? 8 : 0;
  const bf16* p = A + (size_t)(m0 + row) * lda + (k0 + koff);
  v8bf lo = *(const v8bf*)(p);
  v8bf hi = *(const v8bf*)(p + 16);
  v16bf a;
#pragma unroll
  for (int i = 0; i < 8; ++i) { a[i] = lo[i]; a[8 + i] = hi[i]; }
  return a;
}

__device__ __forceinline__ v16bf load_b_frag(const bf16* __restrict__ BT, int ldb,
                                             int n0, int k0, int lane) {
  const int col  = lane & 15;
  const int koff = (lane & 16) ? 8 : 0;
  const bf16* p = BT + (size_t)(n0 + col) * ldb + (k0 + koff);
  v8bf lo = *(const v8bf*)(p);
  v8bf hi = *(const v8bf*)(p + 16);
  v16bf b;
#pragma unroll
  for (int i = 0; i < 8; ++i) { b[i] = lo[i]; b[8 + i] = hi[i]; }
  return b;
}

#define WMMA_BF16(a, b, c) \
  __builtin_amdgcn_wmma_f32_16x16x32_bf16(false, (a), false, (b), (short)0, (c), false, false)

// ---------------------------------------------------------------------------
// Prep kernels: fp32 -> bf16 conversions (weights transposed to [N][K])
// ---------------------------------------------------------------------------
__global__ void cvt_bf16_kernel(const float* __restrict__ src, bf16* __restrict__ dst,
                                size_t n) {
  size_t i = (size_t)blockIdx.x * blockDim.x + threadIdx.x;
  if (i < n) dst[i] = (bf16)src[i];
}

// dst[n*K + k] = bf16(src[k*N + n])   (src is [K][N] row-major)
__global__ void transpose_cvt_kernel(const float* __restrict__ src, bf16* __restrict__ dst,
                                     int K, int N) {
  size_t i = (size_t)blockIdx.x * blockDim.x + threadIdx.x;
  if (i >= (size_t)K * N) return;
  int n = (int)(i / K);
  int k = (int)(i % K);
  dst[i] = (bf16)src[(size_t)k * N + n];
}

__global__ void init_state_kernel(float* __restrict__ Cst, float* __restrict__ Hst,
                                  bf16* __restrict__ Hzero) {
  int i = blockIdx.x * blockDim.x + threadIdx.x;   // B*HD threads
  Cst[i] = 0.f;
  Hst[i] = 0.f;
  Hzero[i] = (bf16)0.f;
}

// ---------------------------------------------------------------------------
// Phase 1: x_proj[t][b][g] = sum_d inp[b][t][d] * Wx[d][g] + bias[g]
// A = inp_bf [B*T][D] (m = b*T + t), BT = WxT [G4][D]. Output bf16 scattered
// to [T][B][G4]. Wave tile: 16 M x 128 N (8 WMMA subtiles). B loads batched
// ahead of the WMMA burst; next A fragment prefetched during compute.
// ---------------------------------------------------------------------------
#define XP_NSUB 8
__global__ void xproj_gemm_kernel(const bf16* __restrict__ A, const bf16* __restrict__ BT,
                                  const float* __restrict__ bias, bf16* __restrict__ xp) {
  const int lane = threadIdx.x & 31;
  const int wave = blockIdx.x * (blockDim.x >> 5) + (threadIdx.x >> 5);
  const int ntiles = G4 / (16 * XP_NSUB);          // 32
  const int m0 = (wave / ntiles) * 16;
  const int n0 = (wave % ntiles) * (16 * XP_NSUB);

  v8f acc[XP_NSUB];
#pragma unroll
  for (int s = 0; s < XP_NSUB; ++s) acc[s] = (v8f){};

  v16bf a = load_a_frag(A, DD, m0, 0, lane);
  for (int k = 0; k < DD; k += 32) {
    v16bf bf[XP_NSUB];
#pragma unroll
    for (int s = 0; s < XP_NSUB; ++s)
      bf[s] = load_b_frag(BT, DD, n0 + s * 16, k, lane);
    v16bf anext = a;
    if (k + 32 < DD) anext = load_a_frag(A, DD, m0, k + 32, lane);
#pragma unroll
    for (int s = 0; s < XP_NSUB; ++s)
      acc[s] = WMMA_BF16(a, bf[s], acc[s]);
    a = anext;
  }

  // C/D layout: lane 0-15 -> N=lane, M=r ; lane 16-31 -> N=lane-16, M=8+r
  const int col   = lane & 15;
  const int rbase = (lane & 16) ? 8 : 0;
#pragma unroll
  for (int s = 0; s < XP_NSUB; ++s) {
    const int n = n0 + s * 16 + col;
    const float bv = bias[n];
#pragma unroll
    for (int r = 0; r < 8; ++r) {
      const int m = m0 + rbase + r;      // m = b*T + t
      const int b_ = m >> 9;             // / TT
      const int t_ = m & (TT - 1);       // % TT
      xp[((size_t)t_ * BB + b_) * G4 + n] = (bf16)(acc[s][r] + bv);
    }
  }
}

// ---------------------------------------------------------------------------
// Phase 2: one timestep.  gates = xp_t + Hprev @ Wh ; LSTM cell update.
//
// Block = 8 waves: 128 b-rows x 16 h-cols x all 4 gates. The 4-gate x 16-col
// x 32-k Wh chunk (4 KB) is moved global->LDS with the CDNA5 async-to-LDS
// path (global_load_async_to_lds_b128, ASYNCcnt), double buffered with a
// padded row stride (40 els -> conflict-free ds_load_b128) and shared by all
// 8 waves. The next chunk's async copy and A fragment are issued before the
// current chunk's 4 WMMAs so memory latency overlaps matrix work.
// ---------------------------------------------------------------------------
#define LSTRIDE 40                        // padded LDS row stride (elements)
__global__ void lstm_step_kernel(const bf16* __restrict__ Hprev,   // [B][HD] bf16
                                 const bf16* __restrict__ WhT,     // [G4][HD] bf16
                                 const bf16* __restrict__ xp_t,    // [B][G4] bf16
                                 float* __restrict__ Cst,          // [B][HD] f32 (in/out)
                                 float* __restrict__ Hst,          // [B][HD] f32 (out)
                                 bf16*  __restrict__ Hall_t) {     // [B][HD] bf16 (out)
  __shared__ bf16 ldsB[2][64 * LSTRIDE];  // 2 x 5 KB

  const int tid   = threadIdx.x;
  const int lane  = tid & 31;
  const int wv    = tid >> 5;             // 0..7
  const int htile = blockIdx.x & 63;      // 64 h tiles
  const int mblk  = blockIdx.x >> 6;      // 0..1
  const int h0 = htile * 16;
  const int b0 = mblk * 128 + wv * 16;

  // cooperative Wh staging: 64 rows (4 gates x 16 cols) x 32 k-els per chunk,
  // one 16-byte async lane-transfer per thread.
  const int ri  = tid >> 2;               // 0..63 : staged row
  const int gi  = ri >> 4;                // gate
  const int cj  = ri & 15;                // col within h tile
  const int seg = (tid & 3) * 8;          // k sub-segment (elements)
  const bf16* gsrc = WhT + (size_t)(gi * HD + h0 + cj) * HD + seg;
  bf16* ldst0 = &ldsB[0][ri * LSTRIDE + seg];
  bf16* ldst1 = &ldsB[1][ri * LSTRIDE + seg];

  // prologue: async-stage chunk k=0 into buffer 0
  async_load_b128_to_lds(gsrc, ldst0);
  wait_asynccnt0();
  __syncthreads();

  v8f acc[4];
#pragma unroll
  for (int g = 0; g < 4; ++g) acc[g] = (v8f){};

  v16bf a = load_a_frag(Hprev, HD, b0, 0, lane);

  const int col   = lane & 15;
  const int koffB = (lane & 16) ? 8 : 0;

  for (int k = 0; k < HD; k += 32) {
    const int  cur  = (k >> 5) & 1;
    const bool more = (k + 32) < HD;

    // issue next chunk's async copy + A-fragment prefetch before computing
    if (more)
      async_load_b128_to_lds(gsrc + (k + 32), cur ? ldst0 : ldst1);
    v16bf anext = a;
    if (more) anext = load_a_frag(Hprev, HD, b0, k + 32, lane);

#pragma unroll
    for (int g = 0; g < 4; ++g) {
      const bf16* p = &ldsB[cur][(g * 16 + col) * LSTRIDE + koffB];
      v8bf lo = *(const v8bf*)(p);
      v8bf hi = *(const v8bf*)(p + 16);
      v16bf b;
#pragma unroll
      for (int i = 0; i < 8; ++i) { b[i] = lo[i]; b[8 + i] = hi[i]; }
      acc[g] = WMMA_BF16(a, b, acc[g]);
    }

    if (more) wait_asynccnt0();
    __syncthreads();
    a = anext;
  }

  // fused LSTM cell update
  const int rbase = (lane & 16) ? 8 : 0;
  const int h = h0 + col;
#pragma unroll
  for (int r = 0; r < 8; ++r) {
    const int b_ = b0 + rbase + r;
    const size_t rowg = (size_t)b_ * G4;
    float gi_ = acc[0][r] + (float)xp_t[rowg + 0 * HD + h];
    float gf_ = acc[1][r] + (float)xp_t[rowg + 1 * HD + h];
    float go_ = acc[2][r] + (float)xp_t[rowg + 2 * HD + h];
    float gc_ = acc[3][r] + (float)xp_t[rowg + 3 * HD + h];

    float I = 1.f / (1.f + __expf(-gi_));
    float F = 1.f / (1.f + __expf(-gf_));
    float O = 1.f / (1.f + __expf(-go_));
    float Ct = tanhf(gc_);

    const size_t idx = (size_t)b_ * HD + h;
    float Cn = F * Cst[idx] + I * Ct;
    float Hn = O * tanhf(Cn);
    Cst[idx] = Cn;
    Hst[idx] = Hn;
    Hall_t[idx] = (bf16)Hn;
  }
}

// ---------------------------------------------------------------------------
// Phase 3: out[b*T + t] = sum_h Hall[t][b][h] * fc_w[h] + fc_b
// One wave per output element, coalesced bf16 loads + shuffle reduction.
// ---------------------------------------------------------------------------
__global__ void fc_kernel(const bf16* __restrict__ Hall, const float* __restrict__ fc_w,
                          const float* __restrict__ fc_b, float* __restrict__ out) {
  const int lane = threadIdx.x & 31;
  const size_t wave = (size_t)blockIdx.x * (blockDim.x >> 5) + (threadIdx.x >> 5); // = t*B + b
  const size_t t = wave / BB;
  const size_t b = wave % BB;
  const bf16* hrow = Hall + wave * HD;
  float s = 0.f;
  for (int h = lane; h < HD; h += 32) s += (float)hrow[h] * fc_w[h];
#pragma unroll
  for (int off = 16; off; off >>= 1) s += __shfl_xor(s, off, 32);
  if (lane == 0) out[b * TT + t] = s + fc_b[0];
}

// ---------------------------------------------------------------------------
// Launcher
// ---------------------------------------------------------------------------
extern "C" void kernel_launch(void* const* d_in, const int* in_sizes, int n_in,
                              void* d_out, int out_size, void* d_ws, size_t ws_size,
                              hipStream_t stream) {
  const float* inp  = (const float*)d_in[0];   // [B][T][D]
  const float* Wx   = (const float*)d_in[1];   // [D][4H]
  const float* Wh   = (const float*)d_in[2];   // [H][4H]
  const float* bias = (const float*)d_in[3];   // [4H]
  const float* fcw  = (const float*)d_in[4];   // [H][1]
  const float* fcb  = (const float*)d_in[5];   // [1]

  float* out  = (float*)d_out;                       // [B][T] = B*T floats
  float* Hout = out + (size_t)BB * TT;               // [B][HD]
  float* Cout = Hout + (size_t)BB * HD;              // [B][HD]

  // workspace carve-out (256B aligned)
  char* ws = (char*)d_ws;
  size_t off = 0;
  auto take = [&](size_t bytes) -> char* {
    char* p = ws + off;
    off = (off + bytes + 255) & ~(size_t)255;
    return p;
  };
  bf16* WhT    = (bf16*)take((size_t)G4 * HD * sizeof(bf16));        //   8 MiB
  bf16* WxT    = (bf16*)take((size_t)G4 * DD * sizeof(bf16));        //   4 MiB
  bf16* inp_bf = (bf16*)take((size_t)BB * TT * DD * sizeof(bf16));   // 128 MiB
  bf16* xp     = (bf16*)take((size_t)TT * BB * G4 * sizeof(bf16));   //   1 GiB
  bf16* Hall   = (bf16*)take((size_t)TT * BB * HD * sizeof(bf16));   // 256 MiB
  bf16* Hzero  = (bf16*)take((size_t)BB * HD * sizeof(bf16));        // 512 KiB

  // --- prep: bf16 conversions ---
  {
    size_t n = (size_t)G4 * DD;       // WxT
    transpose_cvt_kernel<<<(unsigned)((n + 255) / 256), 256, 0, stream>>>(Wx, WxT, DD, G4);
    n = (size_t)G4 * HD;              // WhT
    transpose_cvt_kernel<<<(unsigned)((n + 255) / 256), 256, 0, stream>>>(Wh, WhT, HD, G4);
    n = (size_t)BB * TT * DD;         // inp
    cvt_bf16_kernel<<<(unsigned)((n + 255) / 256), 256, 0, stream>>>(inp, inp_bf, n);
  }
  init_state_kernel<<<(BB * HD) / 256, 256, 0, stream>>>(Cout, Hout, Hzero);

  // --- phase 1: input projection GEMM ---
  {
    const int waves = ((BB * TT) / 16) * (G4 / (16 * XP_NSUB));  // 8192 * 32
    xproj_gemm_kernel<<<waves / 8, 256, 0, stream>>>(inp_bf, WxT, bias, xp);
  }

  // --- phase 2: sequential recurrence (2 m-blocks x 64 h-tiles = 128 blocks) ---
  {
    for (int t = 0; t < TT; ++t) {
      const bf16* Hprev = (t == 0) ? Hzero : (Hall + (size_t)(t - 1) * BB * HD);
      lstm_step_kernel<<<128, 256, 0, stream>>>(
          Hprev, WhT, xp + (size_t)t * BB * G4, Cout, Hout,
          Hall + (size_t)t * BB * HD);
    }
  }

  // --- phase 3: output projection ---
  {
    const size_t waves = (size_t)TT * BB;      // 131072 -> 16384 blocks
    fc_kernel<<<(unsigned)(waves / 8), 256, 0, stream>>>(Hall, fcw, fcb, out);
  }
}
